// Kernel_2837689427019963162_53472342835825
// MI455X (gfx1250) — matrix operand order repaired and checked by reading, not compiled, not run on hardware
//
#include <hip/hip_runtime.h>
#include <hip/hip_bf16.h>

// Problem constants (from reference)
#define NB   32
#define CC   128
#define HH   56
#define WW   56
#define HWSZ 3136   // 56*56
#define K3   384    // C*3
#define CPG  4      // C / G
#define KSPLIT 2
#define KHALF (HWSZ / KSPLIT)   // 1568 = 49 * 32 -> 49 K-steps per wave

#if defined(__has_builtin) && __has_builtin(__builtin_amdgcn_sched_barrier)
#define SCHED_FENCE() __builtin_amdgcn_sched_barrier(0)
#else
#define SCHED_FENCE()
#endif

typedef __bf16 bf16_t;
typedef __bf16 bf16x8  __attribute__((ext_vector_type(8)));
typedef __bf16 bf16x16 __attribute__((ext_vector_type(16)));
typedef float  f32x8   __attribute__((ext_vector_type(8)));

__device__ __forceinline__ bf16_t f2bf(float f) {
  union { float f; unsigned int u; } cv; cv.f = f;
  unsigned int u = cv.u;
  u += 0x7FFFu + ((u >> 16) & 1u);           // round-to-nearest-even
  union { unsigned short s; bf16_t b; } r; r.s = (unsigned short)(u >> 16);
  return r.b;
}

// A-fragment: lane m = l&15; K contiguous kb+16*half..+15
__device__ __forceinline__ bf16x16 loadA(const bf16_t* Arow, int kb, int half) {
  union { bf16x16 v; bf16x8 h[2]; } u;
  u.h[0] = *(const bf16x8*)(Arow + kb + 16 * half);
  u.h[1] = *(const bf16x8*)(Arow + kb + 16 * half + 8);
  return u.v;
}

__device__ __forceinline__ f32x8 bwmma(bf16x16 a, bf16x16 b, f32x8 c) {
  return __builtin_amdgcn_wmma_f32_16x16x32_bf16(false, a, false, b, (short)0, c,
                                                 false, false);
}

// ---------------------------------------------------------------------------
// Prep 1: t2 = p2w * unfold_h(x)  -> bf16 (n, 384, HW)   [A of GEMM1]
//         t6 = roll_h(x,1) + x    -> bf16 (n, 128, HW)   [B of GEMM1, (N,K)]
// ---------------------------------------------------------------------------
__global__ void prep_t2_t6(const float* __restrict__ x,
                           const float* __restrict__ p2w,
                           bf16_t* __restrict__ t2,
                           bf16_t* __restrict__ t6) {
  int idx = blockIdx.x * blockDim.x + threadIdx.x;   // over NB*CC*HWSZ
  if (idx >= NB * CC * HWSZ) return;
  int p = idx % HWSZ;
  int c = (idx / HWSZ) % CC;
  int n = idx / (HWSZ * CC);
  int h = p / WW;
  int w = p % WW;
  const float* xc = x + ((size_t)n * CC + c) * HWSZ;
  float x0  = xc[p];
  float xm2 = (h >= 2)     ? xc[p - 2 * WW] : 0.f;
  float xp2 = (h + 2 < HH) ? xc[p + 2 * WW] : 0.f;
  int   hm1 = (h == 0) ? (HH - 1) : (h - 1);
  float xm1 = xc[hm1 * WW + w];                       // roll(+1) along H
  const float* pw  = p2w + (size_t)c * 3 * HWSZ;
  bf16_t*      t2c = t2 + ((size_t)n * K3 + (size_t)c * 3) * HWSZ;
  t2c[p]            = f2bf(pw[p] * xm2);
  t2c[HWSZ + p]     = f2bf(pw[HWSZ + p] * x0);
  t2c[2 * HWSZ + p] = f2bf(pw[2 * HWSZ + p] * xp2);
  t6[((size_t)n * CC + c) * HWSZ + p] = f2bf(x0 + xm1);
}

// ---------------------------------------------------------------------------
// Prep 2: t3t[p][k] = unfold_w(x)[k][p] -> bf16 (n, HW, 384)  [B of GEMM2]
// (written after GEMM1; aliases t2's buffer)
// ---------------------------------------------------------------------------
__global__ void prep_t3t(const float* __restrict__ x, bf16_t* __restrict__ t3t) {
  int idx = blockIdx.x * blockDim.x + threadIdx.x;   // over NB*HWSZ*K3
  if (idx >= NB * HWSZ * K3) return;
  int k = idx % K3;
  int p = (idx / K3) % HWSZ;
  int n = idx / (K3 * HWSZ);
  int c = k / 3, j = k % 3;
  int w = p % WW;
  int ws = w + 2 * (j - 1);
  float v = 0.f;
  if (ws >= 0 && ws < WW)
    v = x[((size_t)n * CC + c) * HWSZ + (p - w) + ws];
  t3t[idx] = f2bf(v);
}

// ---------------------------------------------------------------------------
// Prep 3: t8[n][o][p] = sum_c conv_w[o][c] * p5w[c] * x[n][c][p]   (4x3136)
// ---------------------------------------------------------------------------
__global__ void prep_t8(const float* __restrict__ x,
                        const float* __restrict__ p5w,
                        const float* __restrict__ conv_w,
                        float* __restrict__ t8) {
  int p = blockIdx.x * blockDim.x + threadIdx.x;
  int n = blockIdx.y;
  if (p >= HWSZ) return;
  float a0 = 0.f, a1 = 0.f, a2 = 0.f, a3 = 0.f;
  const float* xn = x + (size_t)n * CC * HWSZ;
  for (int c = 0; c < CC; ++c) {
    float xv = xn[(size_t)c * HWSZ + p] * p5w[c];
    a0 += conv_w[c] * xv;
    a1 += conv_w[CC + c] * xv;
    a2 += conv_w[2 * CC + c] * xv;
    a3 += conv_w[3 * CC + c] * xv;
  }
  float* o = t8 + (size_t)n * CPG * HWSZ + p;
  o[0] = a0; o[HWSZ] = a1; o[2 * HWSZ] = a2; o[3 * HWSZ] = a3;
}

// ---------------------------------------------------------------------------
// GEMM1: t7[k][c] = sum_p t2[k][p] * t6[c][p]
// M=384, N=128, K=3136. 2-way K split (49 K-steps each). 32x64 strip per
// wave (2 M-tiles x 4 N-tiles = 8 accumulators): halves B-panel re-reads vs
// a 16-row strip and gives 8 WMMA per 12 b128 loads. Ping-pong register
// pipeline with scheduler fences (in-place reloads).
// ---------------------------------------------------------------------------
__global__ void __launch_bounds__(32) gemm1(const bf16_t* __restrict__ t2,
                                            const bf16_t* __restrict__ t6,
                                            float* __restrict__ part) {
  const int lane  = threadIdx.x;
  const int split = blockIdx.z & 1;
  const int n     = blockIdx.z >> 1;
  const int m0    = blockIdx.y * 32;   // 2 k3c tiles
  const int c0    = blockIdx.x * 64;   // channel strip (4 n-tiles)
  const int half  = lane >> 4;
  const int l15   = lane & 15;

  const bf16_t* Arow0 = t2 + (size_t)n * K3 * HWSZ + (size_t)(m0 + l15) * HWSZ;
  const bf16_t* Arow1 = Arow0 + (size_t)16 * HWSZ;
  const bf16_t* Bl    = t6 + (size_t)n * CC * HWSZ +
                        (size_t)(c0 + l15) * HWSZ + 16 * half;

  const int kb0 = split * KHALF;
  f32x8 acc[2][4] = {};

  bf16x16 a0[2], a1[2], b0[4], b1[4];
  a0[0] = loadA(Arow0, kb0, half);
  a0[1] = loadA(Arow1, kb0, half);
#pragma unroll
  for (int t = 0; t < 4; ++t)
    b0[t] = *(const bf16x16*)(Bl + (size_t)(t * 16) * HWSZ + kb0);
  a1[0] = loadA(Arow0, kb0 + 32, half);
  a1[1] = loadA(Arow1, kb0 + 32, half);
#pragma unroll
  for (int t = 0; t < 4; ++t)
    b1[t] = *(const bf16x16*)(Bl + (size_t)(t * 16) * HWSZ + kb0 + 32);

  int kb = kb0;
  for (int i = 0; i < 23; ++i) {         // computes steps 0..45 of 0..48
    __builtin_prefetch(Arow0 + kb + 128, 0, 3);
#pragma unroll
    for (int mt = 0; mt < 2; ++mt)
#pragma unroll
      for (int t = 0; t < 4; ++t) acc[mt][t] = bwmma(a0[mt], b0[t], acc[mt][t]);
    SCHED_FENCE();
    a0[0] = loadA(Arow0, kb + 64, half);
    a0[1] = loadA(Arow1, kb + 64, half);
#pragma unroll
    for (int t = 0; t < 4; ++t)
      b0[t] = *(const bf16x16*)(Bl + (size_t)(t * 16) * HWSZ + kb + 64);
    SCHED_FENCE();
#pragma unroll
    for (int mt = 0; mt < 2; ++mt)
#pragma unroll
      for (int t = 0; t < 4; ++t) acc[mt][t] = bwmma(a1[mt], b1[t], acc[mt][t]);
    SCHED_FENCE();
    a1[0] = loadA(Arow0, kb + 96, half);
    a1[1] = loadA(Arow1, kb + 96, half);
#pragma unroll
    for (int t = 0; t < 4; ++t)
      b1[t] = *(const bf16x16*)(Bl + (size_t)(t * 16) * HWSZ + kb + 96);
    SCHED_FENCE();
    kb += 64;
  }
  // kb = kb0+1472: stage0 = step 46, stage1 = step 47; step 48 still to load
#pragma unroll
  for (int mt = 0; mt < 2; ++mt)
#pragma unroll
    for (int t = 0; t < 4; ++t) acc[mt][t] = bwmma(a0[mt], b0[t], acc[mt][t]);
  SCHED_FENCE();
  a0[0] = loadA(Arow0, kb + 64, half);   // step 48
  a0[1] = loadA(Arow1, kb + 64, half);
#pragma unroll
  for (int t = 0; t < 4; ++t)
    b0[t] = *(const bf16x16*)(Bl + (size_t)(t * 16) * HWSZ + kb + 64);
  SCHED_FENCE();
#pragma unroll
  for (int mt = 0; mt < 2; ++mt)
#pragma unroll
    for (int t = 0; t < 4; ++t) acc[mt][t] = bwmma(a1[mt], b1[t], acc[mt][t]);
#pragma unroll
  for (int mt = 0; mt < 2; ++mt)
#pragma unroll
    for (int t = 0; t < 4; ++t) acc[mt][t] = bwmma(a0[mt], b0[t], acc[mt][t]);

  // partial store, transposed: part[split][n][c][m]
  float* P = part + ((size_t)split * NB + n) * CC * K3;
#pragma unroll
  for (int mt = 0; mt < 2; ++mt)
#pragma unroll
    for (int t = 0; t < 4; ++t) {
      int c = c0 + t * 16 + l15;
      *(f32x8*)(P + (size_t)c * K3 + m0 + mt * 16 + 8 * half) = acc[mt][t];
    }
}

// ---------------------------------------------------------------------------
// Combine the two K-half partials, scale by 1/sqrt(3136)=1/56, emit bf16 t7t.
// Deterministic (fixed two-operand sum), fully coalesced both ways.
// ---------------------------------------------------------------------------
__global__ void combine_t7(const float* __restrict__ part,
                           bf16_t* __restrict__ t7t) {
  int idx = blockIdx.x * blockDim.x + threadIdx.x;   // over NB*CC*K3 (k fastest)
  if (idx >= NB * CC * K3) return;
  float v = part[idx] + part[(size_t)NB * CC * K3 + idx];
  t7t[idx] = f2bf(v * (1.0f / 56.0f));
}

// ---------------------------------------------------------------------------
// GEMM2 + fused epilogue:
//   t10[c][p] = sum_k t7t[c][k] * t3t[p][k] / sqrt(384)
//   out[c][p] = t10[c][p] + t8[c&3][p] * x[c][p]
// M=128, N=3136, K=384 (12 K-steps). 32x64 strip per wave (2 M-tiles x
// 4 N-tiles): halves the 77MB B-panel re-reads. Ping-pong pipe + fences.
// ---------------------------------------------------------------------------
__global__ void __launch_bounds__(32) gemm2(const bf16_t* __restrict__ t7t,
                                            const bf16_t* __restrict__ t3t,
                                            const float* __restrict__ t8,
                                            const float* __restrict__ x,
                                            float* __restrict__ out) {
  const int lane = threadIdx.x;
  const int n    = blockIdx.z;
  const int c0   = blockIdx.y * 32;    // 2 channel tiles
  const int p0   = blockIdx.x * 64;    // 4 pixel tiles
  const int half = lane >> 4;
  const int l15  = lane & 15;

  const bf16_t* Arow0 = t7t + (size_t)n * CC * K3 + (size_t)(c0 + l15) * K3;
  const bf16_t* Arow1 = Arow0 + (size_t)16 * K3;
  const bf16_t* Bl    = t3t + (size_t)n * HWSZ * K3 +
                        (size_t)(p0 + l15) * K3 + 16 * half;

  f32x8 acc[2][4] = {};
  bf16x16 a0[2], a1[2], b0[4], b1[4];
  a0[0] = loadA(Arow0, 0, half);
  a0[1] = loadA(Arow1, 0, half);
#pragma unroll
  for (int t = 0; t < 4; ++t)
    b0[t] = *(const bf16x16*)(Bl + (size_t)(t * 16) * K3);
  a1[0] = loadA(Arow0, 32, half);
  a1[1] = loadA(Arow1, 32, half);
#pragma unroll
  for (int t = 0; t < 4; ++t)
    b1[t] = *(const bf16x16*)(Bl + (size_t)(t * 16) * K3 + 32);

  int kb = 0;
#pragma unroll 1
  for (int i = 0; i < K3 / 64 - 1; ++i) {   // 5 double-steps: steps 0..9
#pragma unroll
    for (int mt = 0; mt < 2; ++mt)
#pragma unroll
      for (int t = 0; t < 4; ++t) acc[mt][t] = bwmma(a0[mt], b0[t], acc[mt][t]);
    SCHED_FENCE();
    a0[0] = loadA(Arow0, kb + 64, half);
    a0[1] = loadA(Arow1, kb + 64, half);
#pragma unroll
    for (int t = 0; t < 4; ++t)
      b0[t] = *(const bf16x16*)(Bl + (size_t)(t * 16) * K3 + kb + 64);
    SCHED_FENCE();
#pragma unroll
    for (int mt = 0; mt < 2; ++mt)
#pragma unroll
      for (int t = 0; t < 4; ++t) acc[mt][t] = bwmma(a1[mt], b1[t], acc[mt][t]);
    SCHED_FENCE();
    a1[0] = loadA(Arow0, kb + 96, half);
    a1[1] = loadA(Arow1, kb + 96, half);
#pragma unroll
    for (int t = 0; t < 4; ++t)
      b1[t] = *(const bf16x16*)(Bl + (size_t)(t * 16) * K3 + kb + 96);
    SCHED_FENCE();
    kb += 64;
  }
#pragma unroll
  for (int mt = 0; mt < 2; ++mt)
#pragma unroll
    for (int t = 0; t < 4; ++t) acc[mt][t] = bwmma(a0[mt], b0[t], acc[mt][t]);
#pragma unroll
  for (int mt = 0; mt < 2; ++mt)
#pragma unroll
    for (int t = 0; t < 4; ++t) acc[mt][t] = bwmma(a1[mt], b1[t], acc[mt][t]);

  const float scale = 0.05103103630798288f;   // 1/sqrt(384)
  const float* xn   = x  + (size_t)n * CC * HWSZ;
  const float* t8n  = t8 + (size_t)n * CPG * HWSZ;
  float*       on   = out + (size_t)n * CC * HWSZ;
#pragma unroll
  for (int mt = 0; mt < 2; ++mt)
#pragma unroll
    for (int t = 0; t < 4; ++t) {
      int p = p0 + t * 16 + l15;
#pragma unroll
      for (int r = 0; r < 8; ++r) {
        int c = c0 + mt * 16 + r + 8 * half;
        on[(size_t)c * HWSZ + p] =
            acc[mt][t][r] * scale +
            t8n[(size_t)(c & 3) * HWSZ + p] * xn[(size_t)c * HWSZ + p];
      }
    }
}

// ---------------------------------------------------------------------------
extern "C" void kernel_launch(void* const* d_in, const int* in_sizes, int n_in,
                              void* d_out, int out_size, void* d_ws, size_t ws_size,
                              hipStream_t stream) {
  const float* x     = (const float*)d_in[0];   // (32,128,56,56)
  const float* p2w   = (const float*)d_in[1];   // (1,128,3,56,56)
  const float* p5w   = (const float*)d_in[2];   // (1,128,1,1)
  const float* convw = (const float*)d_in[3];   // (4,128)
  float* out = (float*)d_out;

  char* ws = (char*)d_ws;
  size_t off = 0;
  auto alloc = [&](size_t bytes) -> void* {
    void* p = ws + off;
    off = (off + bytes + 255) & ~(size_t)255;
    return p;
  };
  bf16_t* t2   = (bf16_t*)alloc((size_t)NB * K3 * HWSZ * 2);        // 77 MB (aliased by t3t)
  bf16_t* t6   = (bf16_t*)alloc((size_t)NB * CC * HWSZ * 2);        // 26 MB
  bf16_t* t7t  = (bf16_t*)alloc((size_t)NB * CC * K3 * 2);          // 3 MB
  float*  t8   = (float*) alloc((size_t)NB * CPG * HWSZ * 4);       // 1.6 MB
  float*  part = (float*) alloc((size_t)KSPLIT * NB * CC * K3 * 4); // 12.6 MB
  bf16_t* t3t  = t2;  // t2 is fully consumed by gemm1 before prep_t3t runs

  {
    int total = NB * CC * HWSZ;
    prep_t2_t6<<<(total + 255) / 256, 256, 0, stream>>>(x, p2w, t2, t6);
  }
  prep_t8<<<dim3((HWSZ + 255) / 256, NB), 256, 0, stream>>>(x, p5w, convw, t8);

  gemm1<<<dim3(CC / 64, K3 / 32, NB * KSPLIT), 32, 0, stream>>>(t2, t6, part);
  {
    int total = NB * CC * K3;
    combine_t7<<<(total + 255) / 256, 256, 0, stream>>>(part, t7t);
  }

  {
    int total = NB * HWSZ * K3;
    prep_t3t<<<(total + 255) / 256, 256, 0, stream>>>(x, t3t);
  }
  gemm2<<<dim3(HWSZ / 64, CC / 32, NB), 32, 0, stream>>>(t7t, t3t, t8, x, out);
}
